// GINModel_48704929137146
// MI455X (gfx1250) — compile-verified, matrix operand order checked by reading, not run on hardware
//
#include <hip/hip_runtime.h>
#include <hip/hip_bf16.h>

// ---------------------------------------------------------------------------
// CDNA5 (gfx1250, wave32) implementation of the GIN + predictor model.
// Dense layers run on v_wmma_f32_16x16x32_f16 (f16 in, f32 accumulate).
// ---------------------------------------------------------------------------

typedef _Float16 half8  __attribute__((ext_vector_type(8)));
typedef _Float16 half16 __attribute__((ext_vector_type(16)));
typedef float    float8 __attribute__((ext_vector_type(8)));

#define IN_CH  128
#define HID    256
#define COMB   384

// ---- x_in = [x | t], and initialize the aggregation accumulator with x_in --
__global__ void k_build_xin(const float* __restrict__ x, const float* __restrict__ t,
                            float* __restrict__ xin, float* __restrict__ hacc, int n) {
    int id = blockIdx.x * blockDim.x + threadIdx.x;
    if (id >= n * IN_CH) return;
    int row = id >> 7, col = id & (IN_CH - 1);
    float v = (col < IN_CH - 1) ? x[(size_t)row * (IN_CH - 1) + col] : t[row];
    xin[id]  = v;
    hacc[id] = v;   // h = x_in + agg  (eps = 0)
}

// ---- segment_sum: one block per edge, 128 lanes = one feature row ----------
__global__ void k_scatter_add(const int* __restrict__ ei, const float* __restrict__ xin,
                              float* __restrict__ hacc, int E) {
    int e = blockIdx.x;
    int f = threadIdx.x;
    int src = ei[e];
    int dst = ei[E + e];
    atomicAdd(&hacc[(size_t)dst * IN_CH + f], xin[(size_t)src * IN_CH + f]);
}

// ---- f32 -> f16 bulk convert ----------------------------------------------
__global__ void k_cvt_f16(const float* __restrict__ in, _Float16* __restrict__ out, int count) {
    int id = blockIdx.x * blockDim.x + threadIdx.x;
    if (id < count) out[id] = (_Float16)in[id];
}

// ---- copy x_in (f16) into columns [256,384) of the concat buffer ----------
__global__ void k_fill_comb(const float* __restrict__ xin, _Float16* __restrict__ comb, int n) {
    int id = blockIdx.x * blockDim.x + threadIdx.x;
    if (id >= n * IN_CH) return;
    int row = id >> 7, col = id & (IN_CH - 1);
    comb[(size_t)row * COMB + HID + col] = (_Float16)xin[id];
}

// ---- transpose + convert weight W[K][C] -> Wt[C][K] in f16 ----------------
__global__ void k_cvt_wt(const float* __restrict__ W, _Float16* __restrict__ Wt, int K, int C) {
    int id = blockIdx.x * blockDim.x + threadIdx.x;
    if (id >= K * C) return;
    int c = id / K, k = id - c * K;
    Wt[id] = (_Float16)W[(size_t)k * C + c];
}

// ---- WMMA f16 operand load: per-lane 2x 16B chunks per documented layout --
// lanes 0-15 : K = kb+0..7  and kb+16..23   (row/col = lane)
// lanes 16-31: K = kb+8..15 and kb+24..31   (row/col = lane-16)
__device__ inline half16 load_frag(const _Float16* __restrict__ p) {
    half8 lo = *(const half8*)(p);
    half8 hi = *(const half8*)(p + 16);
    half16 r;
#pragma unroll
    for (int i = 0; i < 8; ++i) { r[i] = lo[i]; r[i + 8] = hi[i]; }
    return r;
}

template <int ACT>
__device__ inline float apply_act(float v) {
    if (ACT == 0)      { return v > 0.f ? v : 0.f; }
    else if (ACT == 1) { v = tanhf(v); return v > 0.f ? v : 0.f; }
    else               { return v > 0.f ? v : 0.2f * v; }
}

// ---- generic 16x16-tile GEMM with fused bias + activation -----------------
// A   : [rows][LDA]   f16 (row-major)
// Bt  : [ncols][KDIM] f16 (weights pre-transposed: row = output column)
// out : [rows][LDO]   f16
// ACT : 0 = ReLU, 1 = tanh then ReLU, 2 = LeakyReLU(0.2)
// LDA/LDO compile-time so stores become base + immediate offsets.
template <int KDIM, int LDA, int LDO, int ACT>
__global__ __launch_bounds__(32)
void k_gemm16(const _Float16* __restrict__ A,
              const _Float16* __restrict__ Bt,
              const float* __restrict__ bias,
              _Float16* __restrict__ out, int nrows) {
    const int lane = threadIdx.x;
    const int lm   = lane & 15;          // column (B/C) or row (A) within tile
    const int kh   = (lane >> 4) << 3;   // +8 halves for upper half-wave
    const int rowbase = blockIdx.x * 16;
    const int coltile = blockIdx.y;

    int arow_idx = rowbase + lm;
    if (arow_idx >= nrows) arow_idx = nrows - 1;          // tail-safe clamp
    const _Float16* arow = A  + (size_t)arow_idx * LDA + kh;
    const _Float16* brow = Bt + (size_t)(coltile * 16 + lm) * KDIM + kh;

    // C layout: VGPR r, lane l -> row = r + (l>=16 ? 8 : 0), col = l&15.
    // Bias depends only on the column -> replicate across the 8 accum VGPRs.
    float8 c;
    const float bv = bias[coltile * 16 + lm];
#pragma unroll
    for (int r = 0; r < 8; ++r) c[r] = bv;

#pragma unroll
    for (int kb = 0; kb < KDIM; kb += 32) {
        half16 a = load_frag(arow + kb);
        half16 b = load_frag(brow + kb);
        c = __builtin_amdgcn_wmma_f32_16x16x32_f16(
                /*neg_a=*/false, a, /*neg_b=*/false, b,
                /*c_mod=*/(short)0, c, /*reuse_a=*/false, /*reuse_b=*/false);
    }

    const int rofs = (lane >> 4) << 3;   // rows +8 for upper half-wave
    _Float16* op = out + (size_t)(rowbase + rofs) * LDO + coltile * 16 + lm;
    if (rowbase + 16 <= nrows) {
        // full tile: branch-free, stores use constant row-stride offsets
#pragma unroll
        for (int r = 0; r < 8; ++r)
            op[(size_t)r * LDO] = (_Float16)apply_act<ACT>(c[r]);
    } else {
        // tail tile (only when nrows % 16 != 0)
#pragma unroll
        for (int r = 0; r < 8; ++r)
            if (rowbase + rofs + r < nrows)
                op[(size_t)r * LDO] = (_Float16)apply_act<ACT>(c[r]);
    }
}

// ---- final Linear(256 -> 1): one wave per row, shfl reduction -------------
__global__ __launch_bounds__(32)
void k_final(const _Float16* __restrict__ q, const float* __restrict__ w,
             const float* __restrict__ b, float* __restrict__ out, int n) {
    int row  = blockIdx.x;
    int lane = threadIdx.x;
    const half8 v = *(const half8*)(q + (size_t)row * HID + lane * 8);
    float s = 0.f;
#pragma unroll
    for (int i = 0; i < 8; ++i) s += (float)v[i] * w[lane * 8 + i];
#pragma unroll
    for (int m = 16; m >= 1; m >>= 1) s += __shfl_xor(s, m, 32);
    if (lane == 0) {
        out[row]     = 0.f;          // t_pred (reference returns zeros)
        out[n + row] = s + b[0];     // pred
    }
}

extern "C" void kernel_launch(void* const* d_in, const int* in_sizes, int n_in,
                              void* d_out, int out_size, void* d_ws, size_t ws_size,
                              hipStream_t stream) {
    const float* x   = (const float*)d_in[0];
    const float* t   = (const float*)d_in[1];
    /* z = d_in[2] unused by the reference */
    const int*   ei  = (const int*)d_in[3];
    const float* W1  = (const float*)d_in[4];
    const float* b1  = (const float*)d_in[5];
    const float* W2  = (const float*)d_in[6];
    const float* b2  = (const float*)d_in[7];
    const float* P1  = (const float*)d_in[8];
    const float* pb1 = (const float*)d_in[9];
    const float* P2  = (const float*)d_in[10];
    const float* pb2 = (const float*)d_in[11];
    const float* P3  = (const float*)d_in[12];
    const float* pb3 = (const float*)d_in[13];

    const int n = in_sizes[1];          // 50000
    const int E = in_sizes[3] / 2;      // 800000
    const size_t nf = (size_t)n;

    // ---- workspace carve-up (all slices 256B-aligned) ----
    char* ws = (char*)d_ws;
    size_t o = 0;
    float*     xin   = (float*)(ws + o);      o += nf * IN_CH * 4;  // kept until concat
    float*     hacc  = (float*)(ws + o);      o += nf * IN_CH * 4;  // x_in + agg (f32)
    _Float16*  hA    = (_Float16*)(ws + o);   o += nf * IN_CH * 2;  // h in f16
    _Float16*  y1    = (_Float16*)(ws + o);   o += nf * HID  * 2;   // relu(h@W1+b1)
    _Float16*  comb  = (_Float16*)(ws + o);   o += nf * COMB * 2;   // [h2 | x_in] f16
    _Float16*  W1t   = (_Float16*)(ws + o);   o += (size_t)IN_CH * HID * 2;
    _Float16*  W2t   = (_Float16*)(ws + o);   o += (size_t)HID  * HID * 2;
    _Float16*  P1t   = (_Float16*)(ws + o);   o += (size_t)COMB * HID * 2;
    _Float16*  P2t   = (_Float16*)(ws + o);   o += (size_t)HID  * HID * 2;
    // dead-buffer aliasing: xin is free after k_fill_comb, hacc after k_cvt_f16
    _Float16*  q1 = (_Float16*)ws;                       // [N,256] f16 over xin
    _Float16*  q2 = (_Float16*)(ws + nf * IN_CH * 4);    // [N,256] f16 over hacc

    const int tot  = n * IN_CH;
    const int rt   = (n + 15) / 16;     // row tiles (3125)
    const dim3 g2(rt, HID / 16);        // 16 column tiles for all 256-wide layers

    // 1) build x_in and seed accumulator
    k_build_xin<<<(tot + 255) / 256, 256, 0, stream>>>(x, t, xin, hacc, n);
    // 2) segment_sum via f32 atomics (destination is L2-resident: 25.6 MB << 192 MB)
    k_scatter_add<<<E, IN_CH, 0, stream>>>(ei, xin, hacc, E);
    // 3) h -> f16
    k_cvt_f16<<<(tot + 255) / 256, 256, 0, stream>>>(hacc, hA, tot);
    // 4) weights -> transposed f16 (tiny, once per launch)
    k_cvt_wt<<<(IN_CH * HID + 255) / 256, 256, 0, stream>>>(W1, W1t, IN_CH, HID);
    k_cvt_wt<<<(HID  * HID + 255) / 256, 256, 0, stream>>>(W2, W2t, HID,  HID);
    k_cvt_wt<<<(COMB * HID + 255) / 256, 256, 0, stream>>>(P1, P1t, COMB, HID);
    k_cvt_wt<<<(HID  * HID + 255) / 256, 256, 0, stream>>>(P2, P2t, HID,  HID);
    // 5) GIN MLP layer 1: [N,128]@[128,256] + ReLU
    k_gemm16<IN_CH, IN_CH, HID, 0><<<g2, 32, 0, stream>>>(hA, W1t, b1, y1, n);
    // 6) GIN MLP layer 2: [N,256]@[256,256] + tanh + ReLU -> concat cols [0,256)
    k_gemm16<HID, HID, COMB, 1><<<g2, 32, 0, stream>>>(y1, W2t, b2, comb, n);
    // 7) concat cols [256,384) = x_in (f16)
    k_fill_comb<<<(tot + 255) / 256, 256, 0, stream>>>(xin, comb, n);
    // 8) predictor layer 1: [N,384]@[384,256] + LeakyReLU(0.2)
    k_gemm16<COMB, COMB, HID, 2><<<g2, 32, 0, stream>>>(comb, P1t, pb1, q1, n);
    // 9) predictor layer 2: [N,256]@[256,256] + LeakyReLU(0.2)
    k_gemm16<HID, HID, HID, 2><<<g2, 32, 0, stream>>>(q1, P2t, pb2, q2, n);
    // 10) predictor head [N,256]@[256,1] + zero-fill t_pred
    k_final<<<n, 32, 0, stream>>>(q2, P3, pb3, (float*)d_out, n);
}
